// DynamicGraphAttention_22634477650251
// MI455X (gfx1250) — matrix operand order repaired and checked by reading, not compiled, not run on hardware
//
#include <hip/hip_runtime.h>
#include <hip/hip_bf16.h>

// GAT layer for MI455X (gfx1250, wave32, WMMA).
// B=16, L=256, D=128, F_IN=F_OUT=64. One workgroup per (b,l) graph.
// Memory-bound on adj (268 MB) => stream adj/h/out with non-temporal
// cache hints (keep L2 for W/a reused by all 4096 blocks), do both GEMMs
// with v_wmma_f32_16x16x32_f16 out of LDS-staged f16 operands.

#define Bn      16
#define Ln      256
#define Dn      128
#define FIN     64
#define FOUT    64
#define ALPHA   0.2f
#define NEG_INF (-9.0e15f)

#define HS_LD   72    // h tile leading dim (f16), padded: 144B row stride -> conflict-free
#define WT_LD   72    // W^T leading dim (f16)
#define WHT_LD  136   // Wh^T leading dim (f16): 272B row stride -> conflict-free b128
#define ATT_LD  136   // per-wave attention tile leading dim (f16)

typedef __attribute__((ext_vector_type(16))) _Float16 v16h;
typedef __attribute__((ext_vector_type(8)))  _Float16 v8h;
typedef __attribute__((ext_vector_type(4)))  _Float16 v4h;
typedef __attribute__((ext_vector_type(8)))  float    v8f;
typedef __attribute__((ext_vector_type(4)))  float    v4f;
typedef __attribute__((ext_vector_type(4)))  int      v4i;

static __device__ __forceinline__ v16h cat8(v8h lo, v8h hi) {
  return __builtin_shufflevector(lo, hi, 0,1,2,3,4,5,6,7,8,9,10,11,12,13,14,15);
}

__global__ __launch_bounds__(256)
void gat_fused_kernel(const float* __restrict__ h,
                      const int*   __restrict__ adj,
                      const float* __restrict__ W,
                      const float* __restrict__ a,
                      float*       __restrict__ out)
{
  // Stage-1 scratch (h f16 tile + W^T f16) is dead once GEMM1 finishes;
  // reuse it for the 8 per-wave attention tiles in stage 2.
  __shared__ __align__(16) char    scratch[34816];          // max(27648, 34816) B
  __shared__ __align__(16) _Float16 sWhT[FOUT * WHT_LD];    // 17408 B
  __shared__ float sE1[Dn], sE2[Dn];                         // 1024 B
  __shared__ float sA1[FOUT], sA2[FOUT];                     // 512 B

  _Float16* hs   = (_Float16*)scratch;            // [128][HS_LD]
  _Float16* sWT  = hs + Dn * HS_LD;               // [64][WT_LD]  (W transposed: [n][k])
  _Float16* attW = (_Float16*)scratch;            // stage 2: 8 x [16][ATT_LD]

  const int tid  = threadIdx.x;
  const int lane = tid & 31;
  const int wave = tid >> 5;          // 0..7
  const int ln16 = lane & 15;
  const int hi   = lane >> 4;         // 0 or 1 (K/M half select for WMMA frags)
  const size_t blk = blockIdx.x;      // flat (b*L + l)

  // ---- stage W (transposed, f16) and a into LDS (regular-temporal: L2-reused) ----
  for (int idx = tid; idx < FIN * FOUT; idx += 256) {
    const int k = idx >> 6, n = idx & 63;         // W is [k][n]
    sWT[n * WT_LD + k] = (_Float16)W[idx];
  }
  if (tid < FOUT)            sA1[tid]        = a[tid];
  else if (tid < 2 * FOUT)   sA2[tid - FOUT] = a[tid];

  // ---- stage h tile -> f16 LDS (coalesced float4 non-temporal loads) ----
  const float* hBase = h + blk * (size_t)(Dn * FIN);
  for (int idx = tid * 4; idx < Dn * FIN; idx += 256 * 4) {
    const v4f v = __builtin_nontemporal_load((const v4f*)(hBase + idx));
    _Float16* p = hs + (idx >> 6) * HS_LD + (idx & 63);
    p[0] = (_Float16)v.x; p[1] = (_Float16)v.y;
    p[2] = (_Float16)v.z; p[3] = (_Float16)v.w;
  }
  __syncthreads();

  // ================= Stage 1: Wh = h*W, emit Wh^T(f16) + e1,e2 =================
  const int m0 = wave * 16;
  float eiAcc[8], ejAcc[8];
  #pragma unroll
  for (int r = 0; r < 8; ++r) { eiAcc[r] = 0.f; ejAcc[r] = 0.f; }

  #pragma unroll
  for (int n0 = 0; n0 < FOUT; n0 += 16) {
    v8f acc = {};
    #pragma unroll
    for (int k0 = 0; k0 < FIN; k0 += 32) {
      // A frag (16x32 f16): lane = row m0+ln16; halves K {kb..kb+7, kb+16..kb+23}
      const _Float16* ap = hs + (m0 + ln16) * HS_LD + k0 + hi * 8;
      const v16h af = cat8(*(const v8h*)ap, *(const v8h*)(ap + 16));
      // B frag (32x16 f16): lane = col n0+ln16; halves K {kb..kb+7, kb+16..kb+23} in W^T
      const _Float16* bp = sWT + (n0 + ln16) * WT_LD + k0 + hi * 8;
      const v16h bf = cat8(*(const v8h*)bp, *(const v8h*)(bp + 16));
      acc = __builtin_amdgcn_wmma_f32_16x16x32_f16(false, af, false, bf,
                                                   (short)0, acc, false, false);
    }
    // Epilogue: lane holds col n = n0+ln16, rows m = m0 + hi*8 + r (r=0..7).
    const float a1v = sA1[n0 + ln16], a2v = sA2[n0 + ln16];
    v8h wr;
    #pragma unroll
    for (int r = 0; r < 8; ++r) {
      const float v = acc[r];
      eiAcc[r] += v * a1v;
      ejAcc[r] += v * a2v;
      wr[r] = (_Float16)v;
    }
    // Transposed store: Wh^T[n][m0+hi*8 .. +8] is contiguous -> one ds_store_b128.
    *(v8h*)(sWhT + (n0 + ln16) * WHT_LD + m0 + hi * 8) = wr;
  }

  // Reduce e_i/e_j over the 16 lanes sharing each row (xor masks never cross hi bit).
  #pragma unroll
  for (int r = 0; r < 8; ++r) {
    float ei = eiAcc[r], ej = ejAcc[r];
    #pragma unroll
    for (int s = 1; s < 16; s <<= 1) {
      ei += __shfl_xor(ei, s);
      ej += __shfl_xor(ej, s);
    }
    if (ln16 == 0) {
      sE1[m0 + hi * 8 + r] = ei;
      sE2[m0 + hi * 8 + r] = ej;
    }
  }
  __syncthreads();   // stage-1 scratch now dead; attW may overwrite it

  // ================= Stage 2: masked softmax rows + h' = att*Wh =================
  const int i0 = wave * 16;
  _Float16* att = attW + wave * (16 * ATT_LD);
  const int* adjBase = adj + blk * (size_t)(Dn * Dn) + (size_t)i0 * Dn + lane * 4;
  const v4f e2v = *(const v4f*)(sE2 + lane * 4);   // this lane's 4 columns

  #pragma unroll 4   // interleave independent rows: hide shuffle/exp latency
  for (int row = 0; row < 16; ++row) {
    // adj is a pure one-touch stream: non-temporal so it doesn't thrash L2.
    const v4i am = __builtin_nontemporal_load((const v4i*)(adjBase + row * Dn));
    const float ei = sE1[i0 + row];
    float s[4];
    {
      float t;
      t = ei + e2v.x; t = t > 0.f ? t : ALPHA * t; s[0] = am.x > 0 ? t : NEG_INF;
      t = ei + e2v.y; t = t > 0.f ? t : ALPHA * t; s[1] = am.y > 0 ? t : NEG_INF;
      t = ei + e2v.z; t = t > 0.f ? t : ALPHA * t; s[2] = am.z > 0 ? t : NEG_INF;
      t = ei + e2v.w; t = t > 0.f ? t : ALPHA * t; s[3] = am.w > 0 ? t : NEG_INF;
    }
    float mx = fmaxf(fmaxf(s[0], s[1]), fmaxf(s[2], s[3]));
    #pragma unroll
    for (int d = 1; d < 32; d <<= 1) mx = fmaxf(mx, __shfl_xor(mx, d));
    float p[4], sum = 0.f;
    #pragma unroll
    for (int c = 0; c < 4; ++c) { p[c] = __expf(s[c] - mx); sum += p[c]; }
    #pragma unroll
    for (int d = 1; d < 32; d <<= 1) sum += __shfl_xor(sum, d);
    const float inv = 1.0f / sum;
    v4h q;
    #pragma unroll
    for (int c = 0; c < 4; ++c) q[c] = (_Float16)(p[c] * inv);
    *(v4h*)(att + row * ATT_LD + lane * 4) = q;   // ds_store_b64, same-wave in-order
  }

  // att(16x128) @ Wh(128x64) via 16 WMMAs per wave.
  float* outBase = out + blk * (size_t)(Dn * FOUT);
  #pragma unroll
  for (int n0 = 0; n0 < FOUT; n0 += 16) {
    v8f acc = {};
    #pragma unroll
    for (int k0 = 0; k0 < Dn; k0 += 32) {
      const _Float16* ap = att + ln16 * ATT_LD + k0 + hi * 8;
      const v16h af = cat8(*(const v8h*)ap, *(const v8h*)(ap + 16));
      const _Float16* bp = sWhT + (n0 + ln16) * WHT_LD + k0 + hi * 8;
      const v16h bf = cat8(*(const v8h*)bp, *(const v8h*)(bp + 16));
      acc = __builtin_amdgcn_wmma_f32_16x16x32_f16(false, af, false, bf,
                                                   (short)0, acc, false, false);
    }
    // out is write-once: non-temporal stores, coalesced per 16-lane group.
    #pragma unroll
    for (int r = 0; r < 8; ++r)
      __builtin_nontemporal_store(acc[r],
          outBase + (size_t)(i0 + hi * 8 + r) * FOUT + n0 + ln16);
  }
}

extern "C" void kernel_launch(void* const* d_in, const int* in_sizes, int n_in,
                              void* d_out, int out_size, void* d_ws, size_t ws_size,
                              hipStream_t stream) {
  (void)in_sizes; (void)n_in; (void)out_size; (void)d_ws; (void)ws_size;
  const float* h   = (const float*)d_in[0];
  const int*   adj = (const int*)d_in[1];
  const float* W   = (const float*)d_in[2];
  const float* a   = (const float*)d_in[3];
  float*       out = (float*)d_out;
  dim3 grid(Bn * Ln);   // 4096 workgroups, one per (b,l) graph
  dim3 block(256);      // 8 wave32 waves
  gat_fused_kernel<<<grid, block, 0, stream>>>(h, adj, W, a, out);
}
